// Hidden_Blocks_46969762349152
// MI455X (gfx1250) — matrix operand order repaired and checked by reading, not compiled, not run on hardware
//
#include <hip/hip_runtime.h>

#define N_DIM 4096
#define NL    8
#define EPSV  1e-5f

typedef __attribute__((ext_vector_type(16))) __bf16 v16bf;
typedef __attribute__((ext_vector_type(8)))  float  v8f;
typedef __attribute__((ext_vector_type(4)))  int    v4i;

typedef __attribute__((address_space(1))) v4i* gv4i_ptr;
typedef __attribute__((address_space(3))) v4i* lv4i_ptr;

// ---- CDNA5 feature detection (compile-probe friendly) ----
#if defined(__has_builtin)
#  if __has_builtin(__builtin_amdgcn_global_load_async_to_lds_b128)
#    define ASYNC_OK 1
#  endif
#  if __has_builtin(__builtin_amdgcn_s_wait_asynccnt)
#    define HAVE_WAITASYNC 1
#  endif
#endif
#ifndef ASYNC_OK
#  define ASYNC_OK 0
#endif
#ifndef HAVE_WAITASYNC
#  define HAVE_WAITASYNC 0
#endif

__device__ __forceinline__ void wait_async() {
#if ASYNC_OK
#  if HAVE_WAITASYNC
    __builtin_amdgcn_s_wait_asynccnt(0);
#  else
    asm volatile("s_wait_asynccnt 0" ::: "memory");
#  endif
#endif
}

// one 16B async (or sync fallback) copy global -> LDS
__device__ __forceinline__ void cp16(const unsigned short* gp, unsigned short* lp) {
#if ASYNC_OK
    __builtin_amdgcn_global_load_async_to_lds_b128((gv4i_ptr)gp, (lv4i_ptr)lp, 0, 0);
#else
    *(uint4*)lp = *(const uint4*)gp;
#endif
}

// pack two fp32 -> packed bf16x2 (RNE) — used only in bandwidth-bound kernels
__device__ __forceinline__ unsigned int pack_bf16(float lo, float hi) {
    unsigned int ul = __builtin_bit_cast(unsigned int, lo);
    unsigned int uh = __builtin_bit_cast(unsigned int, hi);
    ul = ((ul + 0x7FFFu + ((ul >> 16) & 1u)) >> 16) & 0xFFFFu;
    uh = ((uh + 0x7FFFu + ((uh >> 16) & 1u)) >> 16) & 0xFFFFu;
    return ul | (uh << 16);
}

// =====================================================================
// Kernel 0: streaming fp32 -> bf16 weight conversion (one layer, 16M elems)
// =====================================================================
__global__ __launch_bounds__(256) void wcvt_kernel(
    const float* __restrict__ W, unsigned short* __restrict__ Wbf)
{
    const size_t base = ((size_t)blockIdx.x * 256 + threadIdx.x) * 8;
    float4 a = ((const float4*)(W + base))[0];
    float4 b = ((const float4*)(W + base))[1];
    uint4 o;
    o.x = pack_bf16(a.x, a.y);
    o.y = pack_bf16(a.z, a.w);
    o.z = pack_bf16(b.x, b.y);
    o.w = pack_bf16(b.z, b.w);
    *(uint4*)(Wbf + base) = o;
}

// =====================================================================
// Kernel 1: fused LayerNorm -> bf16 activation buffer
// =====================================================================
__global__ __launch_bounds__(256) void ln_to_bf16_kernel(
    const float* __restrict__ x, const float* __restrict__ g,
    const float* __restrict__ bt, unsigned short* __restrict__ out)
{
    const int row = blockIdx.x;
    const int t   = threadIdx.x;
    const float* xr = x + (size_t)row * N_DIM;

    float4 v[4];
    float s = 0.f, q = 0.f;
#pragma unroll
    for (int i = 0; i < 4; ++i) {
        v[i] = ((const float4*)xr)[t * 4 + i];
        s += v[i].x + v[i].y + v[i].z + v[i].w;
        q += v[i].x * v[i].x + v[i].y * v[i].y + v[i].z * v[i].z + v[i].w * v[i].w;
    }
#pragma unroll
    for (int off = 16; off > 0; off >>= 1) {
        s += __shfl_xor(s, off, 32);
        q += __shfl_xor(q, off, 32);
    }
    __shared__ float rs[8], rq[8];
    if ((t & 31) == 0) { rs[t >> 5] = s; rq[t >> 5] = q; }
    __syncthreads();
    s = 0.f; q = 0.f;
#pragma unroll
    for (int i = 0; i < 8; ++i) { s += rs[i]; q += rq[i]; }

    const float mu  = s * (1.f / N_DIM);
    const float var = q * (1.f / N_DIM) - mu * mu;
    const float inv = rsqrtf(var + EPSV);

    unsigned int ui[8];
#pragma unroll
    for (int i = 0; i < 4; ++i) {
        float4 gv = ((const float4*)g)[t * 4 + i];
        float4 bv = ((const float4*)bt)[t * 4 + i];
        float e0 = (v[i].x - mu) * inv * gv.x + bv.x;
        float e1 = (v[i].y - mu) * inv * gv.y + bv.y;
        float e2 = (v[i].z - mu) * inv * gv.z + bv.z;
        float e3 = (v[i].w - mu) * inv * gv.w + bv.w;
        ui[i * 2]     = pack_bf16(e0, e1);
        ui[i * 2 + 1] = pack_bf16(e2, e3);
    }
    uint4* o = (uint4*)(out + (size_t)row * N_DIM + t * 16);
    o[0] = make_uint4(ui[0], ui[1], ui[2], ui[3]);
    o[1] = make_uint4(ui[4], ui[5], ui[6], ui[7]);
}

// =====================================================================
// Kernel 2: Y = lnA(bf16) @ Wbf^T + bias, then per-feature LeakyReLU.
// 128x128 tile / 256 threads, BK=32, double-buffered LDS (flat arrays,
// dynamic buffer offset, K-loop NOT unrolled to keep accumulators pinned),
// both tiles via async global->LDS, bf16 WMMA, f32 accumulate.
// =====================================================================
#define BB   128
#define BM   128
#define BK   32
#define NKT  (N_DIM / BK)
#define TILE (BB * BK)          // 4096 ushorts per buffer

__global__ __launch_bounds__(256) void gemm_bias_lrelu_kernel(
    const unsigned short* __restrict__ A,   // [BATCH][N] bf16 (LN output)
    const unsigned short* __restrict__ Wb,  // [M][K] bf16 row-major
    const float* __restrict__ bias,         // [N]
    const float* __restrict__ slope,        // [N]
    float* __restrict__ out)                // [BATCH][N] fp32
{
    __shared__ unsigned short lA[2 * TILE];   // 16 KB
    __shared__ unsigned short lB[2 * TILE];   // 16 KB ([feature][k])

    const int t    = threadIdx.x;
    const int lane = t & 31;
    const int w    = t >> 5;
    const int ln16 = lane & 15;
    const int lh   = lane >> 4;

    const int mBase = blockIdx.x * BM;
    const int bBase = blockIdx.y * BB;

    const int wave_b = w & 1;           // 2 waves x 64 batch rows
    const int wave_m = w >> 1;          // 4 waves x 32 features

    v8f acc[4][2] = {};

    // staging: 512 x 16B transfers per tile, 2 per thread (rows r0, r0+64)
    const int r0 = t >> 2, sseg = (t & 3) * 8;
    const unsigned short* gA0 = A  + (size_t)(bBase + r0) * N_DIM + sseg;
    const unsigned short* gA1 = gA0 + (size_t)64 * N_DIM;
    const unsigned short* gB0 = Wb + (size_t)(mBase + r0) * N_DIM + sseg;
    const unsigned short* gB1 = gB0 + (size_t)64 * N_DIM;
    const int l0 = r0 * BK + sseg;
    const int l1 = l0 + 64 * BK;

    // per-lane fragment base offsets (ushort units) inside one buffer
    // A (16x32 MxK): lanes 0-15 -> K=0..7 & 16..23 ; lanes 16-31 -> K=8..15 & 24..31
    const int aBase = (wave_b * 64 + ln16) * BK + lh * 8;
    // B (32x16 KxN): lanes 0-15 -> K=0..7 & 16..23 ; lanes 16-31 -> K=8..15 & 24..31
    const int bBase2 = (wave_m * 32 + ln16) * BK + lh * 8;

    auto stage = [&](int off, int kt) {
        const int ko = kt * BK;
        cp16(gA0 + ko, &lA[off + l0]);
        cp16(gA1 + ko, &lA[off + l1]);
        cp16(gB0 + ko, &lB[off + l0]);
        cp16(gB1 + ko, &lB[off + l1]);
    };

    auto compute = [&](int off) {
        union FragU { uint4 q[2]; v16bf v; };
        FragU a[4], bf[2];
#pragma unroll
        for (int i = 0; i < 4; ++i) {
            const unsigned short* p = &lA[off + aBase + i * 16 * BK];
            a[i].q[0] = *(const uint4*)p;
            a[i].q[1] = *(const uint4*)(p + 16);
        }
#pragma unroll
        for (int j = 0; j < 2; ++j) {
            const unsigned short* p = &lB[off + bBase2 + j * 16 * BK];
            bf[j].q[0] = *(const uint4*)p;
            bf[j].q[1] = *(const uint4*)(p + 16);
        }
#pragma unroll
        for (int i = 0; i < 4; ++i)
#pragma unroll
            for (int j = 0; j < 2; ++j)
                acc[i][j] = __builtin_amdgcn_wmma_f32_16x16x32_bf16(
                    false, a[i].v, false, bf[j].v, (short)0, acc[i][j], false, false);
    };

    // prologue: stage tile 0 into buffer 0
    stage(0, 0);
    wait_async();
    __syncthreads();

    // IMPORTANT: no unrolling — keeps one loop body so accumulators stay
    // pinned in one register assignment (no v_mov_b64 shuffles between
    // unrolled copies), buffer select is a cheap dynamic offset.
#pragma clang loop unroll(disable)
    for (int kt = 0; kt < NKT; ++kt) {
        const int cur = (kt & 1) * TILE;
        const int nxt = TILE - cur;
        if (kt + 1 < NKT) stage(nxt, kt + 1);   // overlap copy(k+1) with compute(k)
        compute(cur);
        wait_async();
        __syncthreads();
    }

    // epilogue: bias + per-feature LeakyReLU, fp32 out
#pragma unroll
    for (int j = 0; j < 2; ++j) {
        const int col = mBase + wave_m * 32 + j * 16 + ln16;  // D: N = lane%16
        const float bj = bias[col];
        const float sj = slope[col];
#pragma unroll
        for (int i = 0; i < 4; ++i) {
#pragma unroll
            for (int r = 0; r < 8; ++r) {
                const int rowm = bBase + wave_b * 64 + i * 16 + r + 8 * lh;  // D: M = r + 8*(lane/16)
                float v2 = acc[i][j][r] + bj;
                v2 = (v2 < 0.f) ? v2 * sj : v2;
                out[(size_t)rowm * N_DIM + col] = v2;
            }
        }
    }
}

// =====================================================================
// Host launcher: 8 chained layers, ping-pong activations through d_ws.
// ws layout: [buf0 f32 64MB][buf1 f32 64MB][lnbuf bf16 32MB][wbf bf16 32MB]
// =====================================================================
extern "C" void kernel_launch(void* const* d_in, const int* in_sizes, int n_in,
                              void* d_out, int out_size, void* d_ws, size_t ws_size,
                              hipStream_t stream) {
    const float* x     = (const float*)d_in[0];
    const float* ln_g  = (const float*)d_in[1];
    const float* ln_b  = (const float*)d_in[2];
    const float* W     = (const float*)d_in[3];
    const float* b     = (const float*)d_in[4];
    const float* slope = (const float*)d_in[5];
    float* out = (float*)d_out;

    const size_t NE = (size_t)N_DIM * N_DIM;
    float* buf0 = (float*)d_ws;
    float* buf1 = buf0 + NE;
    unsigned short* lnbuf = (unsigned short*)(buf1 + NE);
    unsigned short* wbf   = lnbuf + NE;

    for (int i = 0; i < NL; ++i) {
        const float* src = (i == 0) ? x : ((i & 1) ? buf0 : buf1);
        float* dst = (i == NL - 1) ? out : ((i & 1) ? buf1 : buf0);

        wcvt_kernel<<<(int)(NE / (256 * 8)), 256, 0, stream>>>(W + (size_t)i * NE, wbf);

        ln_to_bf16_kernel<<<N_DIM, 256, 0, stream>>>(
            src, ln_g + (size_t)i * N_DIM, ln_b + (size_t)i * N_DIM, lnbuf);

        gemm_bias_lrelu_kernel<<<dim3(N_DIM / BM, N_DIM / BB), 256, 0, stream>>>(
            lnbuf, wbf, b + (size_t)i * N_DIM,
            slope + (size_t)i * N_DIM, dst);
    }
}
